// SequenceLSTM_19748259627557
// MI455X (gfx1250) — matrix operand order repaired and checked by reading, not compiled, not run on hardware
//
#include <hip/hip_runtime.h>
#include <hip/hip_bf16.h>

// Problem constants (fixed by the reference).
#define B_SZ    1024
#define T_MAIN  1000
#define H       51
#define FUT     200
#define TT      (T_MAIN + FUT)   // 1200
#define HP      64               // padded hidden (K granularity)
#define NP      256              // 4 gates * HP padded output cols
#define MROWS   16               // batch rows per block = WMMA M
#define NTHREADS 128             // 4 wave32's; wave w owns hidden tile j in [16w,16w+16)

typedef __attribute__((ext_vector_type(16))) __bf16 bf16x16;
typedef __attribute__((ext_vector_type(8)))  float  f32x8;

union BVec { bf16x16 v; float4 f4[2]; };

__device__ __forceinline__ unsigned short f2bf(float f) {
  unsigned u = __float_as_uint(f);
  u = (u + 0x7FFFu + ((u >> 16) & 1u)) >> 16;   // RNE f32 -> bf16
  return (unsigned short)u;
}
__device__ __forceinline__ float sigmoidf_(float x) {
  x = fminf(fmaxf(x, -30.f), 30.f);
  float e = __expf(-x);
  return __builtin_amdgcn_rcpf(1.f + e);
}
__device__ __forceinline__ float tanhf_(float x) {
  x = fminf(fmaxf(x, -15.f), 15.f);
  float e = __expf(-2.f * x);
  return (1.f - e) * __builtin_amdgcn_rcpf(1.f + e);
}
__device__ __forceinline__ f32x8 splat8(float v) {
  f32x8 r;
  #pragma unroll
  for (int i = 0; i < 8; ++i) r[i] = v;
  return r;
}
__device__ __forceinline__ f32x8 wmma_bf16(const BVec& a, const BVec& b, f32x8 c) {
  // D = A(16x32 bf16) * B(32x16 bf16) + C(16x16 f32)
  return __builtin_amdgcn_wmma_f32_16x16x32_bf16(false, a.v, false, b.v,
                                                 (short)0, c, false, false);
}
// A-operand layout: lane l<16 -> M=l, K = kb+{0..7, 16..23}; lane>=16 -> K = kb+{8..15, 24..31}
__device__ __forceinline__ void loadA(BVec& a, const unsigned short* hp,
                                      int kb, int lm, int hi) {
  const unsigned short* p = hp + lm * HP + kb + (hi ? 8 : 0);
  a.f4[0] = *(const float4*)p;
  a.f4[1] = *(const float4*)(p + 16);
}
// B tiles pre-packed so each lane reads 32 contiguous bytes.
__device__ __forceinline__ void loadB(BVec& b, const unsigned int* wp, int bi, int lane) {
  const float4* p = (const float4*)(wp + (bi << 8) + (lane << 3));
  b.f4[0] = p[0];
  b.f4[1] = p[1];
}

// Padded weight W[layer][k][n]: n = gate*64 + jj (jj<51 valid), gate rows in global = gate*51+jj.
// layer0: K=64  (k<51: w_hh0[row][k];  k==51: w_ih0[row] (x folded in);  else 0)
// layer1: K=128 (k<51: w_ih1[row][k];  64<=k<115: w_hh1[row][k-64];     else 0)
__device__ __forceinline__ float fetchW(int layer, int k, int n,
    const float* w_ih0, const float* w_hh0,
    const float* w_ih1, const float* w_hh1) {
  int g = n >> 6, jj = n & 63;
  if (jj >= H) return 0.f;
  int row = g * H + jj;
  if (layer == 0) {
    if (k < H)  return w_hh0[row * H + k];
    if (k == H) return w_ih0[row];
    return 0.f;
  } else {
    if (k < H) return w_ih1[row * H + k];
    if (k >= HP && k < HP + H) return w_hh1[row * H + (k - HP)];
    return 0.f;
  }
}

__global__ __launch_bounds__(NTHREADS)
void SequenceLSTM_kernel(
    const float* __restrict__ input,
    const float* __restrict__ w_ih0, const float* __restrict__ w_hh0,
    const float* __restrict__ b_ih0, const float* __restrict__ b_hh0,
    const float* __restrict__ w_ih1, const float* __restrict__ w_hh1,
    const float* __restrict__ b_ih1, const float* __restrict__ b_hh1,
    const float* __restrict__ w_lin, const float* __restrict__ b_lin,
    float* __restrict__ out) {
  // 96 B-layout tile blocks (L0: 2 kchunks x 16 ntiles; L1: 4 x 16), 1KB each.
  __shared__ __align__(16) unsigned int   wpack[96 * 256];     // 96 KB bf16-packed weights
  __shared__ float bias0s[NP];
  __shared__ float bias1s[NP];
  __shared__ float wlins[HP];
  __shared__ float blins;
  __shared__ __align__(16) unsigned short hp0[MROWS * HP];     // h0 (bf16, A-source), slot k=51 = x
  __shared__ __align__(16) unsigned short hp1[MROWS * HP];     // h1 (bf16, A-source)
  __shared__ __align__(16) float          hf1[MROWS * HP];     // h1 (f32, for linear head)
  __shared__ float xbuf[MROWS];                                // fed-back x for future phase

  const int tid  = threadIdx.x;
  const int wv   = tid >> 5;        // wave id = hidden j-tile index (0..3)
  const int lane = tid & 31;
  const int lm   = lane & 15;
  const int hi   = lane >> 4;       // 0: rows 0..7 / K-low half, 1: rows 8..15 / K-high half
  const int m0   = blockIdx.x * MROWS;
  const int jcol = wv * 16 + lm;    // this lane's hidden unit index (D-layout column)

  // ---- one-time pack: weights -> WMMA B layout (bf16 pairs per dword) ----
  for (int idx = tid; idx < 96 * 256; idx += NTHREADS) {
    int bi = idx >> 8, rem = idx & 255;
    int l = rem >> 3, v = rem & 7;
    int layer, kc, nt;
    if (bi < 32) { layer = 0; kc = bi >> 4;        nt = bi & 15; }
    else         { layer = 1; kc = (bi - 32) >> 4; nt = (bi - 32) & 15; }
    int k0 = kc * 32 + ((l < 16) ? 0 : 8) + 2 * v + ((v < 4) ? 0 : 8);
    int n  = nt * 16 + (l & 15);
    float wa = fetchW(layer, k0,     n, w_ih0, w_hh0, w_ih1, w_hh1);
    float wb = fetchW(layer, k0 + 1, n, w_ih0, w_hh0, w_ih1, w_hh1);
    wpack[idx] = (unsigned)f2bf(wa) | ((unsigned)f2bf(wb) << 16);
  }
  for (int n = tid; n < NP; n += NTHREADS) {
    int g = n >> 6, jj = n & 63;
    float v0 = 0.f, v1 = 0.f;
    if (jj < H) {
      v0 = b_ih0[g * H + jj] + b_hh0[g * H + jj];
      v1 = b_ih1[g * H + jj] + b_hh1[g * H + jj];
    }
    bias0s[n] = v0; bias1s[n] = v1;
  }
  for (int i = tid; i < MROWS * HP; i += NTHREADS) { hp0[i] = 0; hp1[i] = 0; }
  if (tid < HP)  wlins[tid] = (tid < H) ? w_lin[tid] : 0.f;
  if (tid == 0)  blins = b_lin[0];
  __syncthreads();

  float br0[4], br1[4];
  #pragma unroll
  for (int g = 0; g < 4; ++g) {
    br0[g] = bias0s[g * 64 + jcol];
    br1[g] = bias1s[g * 64 + jcol];
  }
  f32x8 c0 = splat8(0.f), c1 = splat8(0.f);   // cell states, D-layout resident

  for (int t = 0; t < TT; ++t) {
    __syncthreads();                                   // B0: prev hp1/xbuf writes visible
    if (tid < MROWS) {
      float x = (t < T_MAIN) ? input[(m0 + tid) * T_MAIN + t] : xbuf[tid];
      hp0[tid * HP + H] = f2bf(x);                     // fold x into K-slot 51
    }
    __syncthreads();                                   // B1

    BVec a0lo, a0hi, a1l, a1h;
    loadA(a0lo, hp0, 0, lm, hi);  loadA(a0hi, hp0, 32, lm, hi);
    loadA(a1l,  hp1, 0, lm, hi);  loadA(a1h,  hp1, 32, lm, hi);

    // ---- layer 0: gates = [x|h0] @ W0, tiles {g*4+wv} ----
    f32x8 acc[4];
    #pragma unroll
    for (int g = 0; g < 4; ++g) {
      int gt = g * 4 + wv;
      f32x8 a = splat8(br0[g]);
      BVec b;
      loadB(b, wpack, 0 * 16 + gt, lane); a = wmma_bf16(a0lo, b, a);
      loadB(b, wpack, 1 * 16 + gt, lane); a = wmma_bf16(a0hi, b, a);
      acc[g] = a;
    }
    float h0v[8];
    #pragma unroll
    for (int r = 0; r < 8; ++r) {
      float ii = sigmoidf_(acc[0][r]);
      float ff = sigmoidf_(acc[1][r]);
      float gg = tanhf_(acc[2][r]);
      float oo = sigmoidf_(acc[3][r]);
      float cn = ff * c0[r] + ii * gg;
      c0[r] = cn;
      h0v[r] = oo * tanhf_(cn);
    }
    __syncthreads();                                   // B2a: all waves done reading hp0/hp1
    #pragma unroll
    for (int r = 0; r < 8; ++r)
      hp0[(r + (hi << 3)) * HP + jcol] = f2bf(h0v[r]);
    __syncthreads();                                   // B2: new h0 visible

    BVec a0nlo, a0nhi;
    loadA(a0nlo, hp0, 0, lm, hi);  loadA(a0nhi, hp0, 32, lm, hi);

    // ---- layer 1: gates = [h0_new | h1] @ W1 (K=128) ----
    #pragma unroll
    for (int g = 0; g < 4; ++g) {
      int gt = g * 4 + wv;
      f32x8 a = splat8(br1[g]);
      BVec b;
      loadB(b, wpack, 32 + 0 * 16 + gt, lane); a = wmma_bf16(a0nlo, b, a);
      loadB(b, wpack, 32 + 1 * 16 + gt, lane); a = wmma_bf16(a0nhi, b, a);
      loadB(b, wpack, 32 + 2 * 16 + gt, lane); a = wmma_bf16(a1l,  b, a);
      loadB(b, wpack, 32 + 3 * 16 + gt, lane); a = wmma_bf16(a1h,  b, a);
      acc[g] = a;
    }
    #pragma unroll
    for (int r = 0; r < 8; ++r) {
      float ii = sigmoidf_(acc[0][r]);
      float ff = sigmoidf_(acc[1][r]);
      float gg = tanhf_(acc[2][r]);
      float oo = sigmoidf_(acc[3][r]);
      float cn = ff * c1[r] + ii * gg;
      c1[r] = cn;
      float h = oo * tanhf_(cn);
      int m = r + (hi << 3);
      hp1[m * HP + jcol] = f2bf(h);
      hf1[m * HP + jcol] = h;
    }
    __syncthreads();                                   // B3: h1 visible for head

    if (tid < MROWS) {                                 // linear head + feedback
      float s = blins;
      const float* hrow = &hf1[tid * HP];
      #pragma unroll
      for (int j = 0; j < H; ++j) s += hrow[j] * wlins[j];
      out[(m0 + tid) * TT + t] = s;
      xbuf[tid] = s;
    }
  }
}

extern "C" void kernel_launch(void* const* d_in, const int* in_sizes, int n_in,
                              void* d_out, int out_size, void* d_ws, size_t ws_size,
                              hipStream_t stream) {
  (void)in_sizes; (void)n_in; (void)out_size; (void)d_ws; (void)ws_size;
  const float* input = (const float*)d_in[0];
  const float* w_ih0 = (const float*)d_in[1];
  const float* w_hh0 = (const float*)d_in[2];
  const float* b_ih0 = (const float*)d_in[3];
  const float* b_hh0 = (const float*)d_in[4];
  const float* w_ih1 = (const float*)d_in[5];
  const float* w_hh1 = (const float*)d_in[6];
  const float* b_ih1 = (const float*)d_in[7];
  const float* b_hh1 = (const float*)d_in[8];
  const float* w_lin = (const float*)d_in[9];
  const float* b_lin = (const float*)d_in[10];
  float* out = (float*)d_out;

  dim3 grid(B_SZ / MROWS);   // 64 blocks, each owns 16 batch rows for all 1200 steps
  dim3 block(NTHREADS);      // 4 wave32's: wave w owns hidden tile j in [16w, 16w+16)
  hipLaunchKernelGGL(SequenceLSTM_kernel, grid, block, 0, stream,
                     input, w_ih0, w_hh0, b_ih0, b_hh0,
                     w_ih1, w_hh1, b_ih1, b_hh1, w_lin, b_lin, out);
}
